// SinkhornAttention_52123723104830
// MI455X (gfx1250) — matrix operand order repaired and checked by reading, not compiled, not run on hardware
//
#include <hip/hip_runtime.h>

typedef _Float16 half_t;
typedef __attribute__((ext_vector_type(16))) _Float16 v16h;
typedef __attribute__((ext_vector_type(8)))  float    v8f;
typedef __attribute__((ext_vector_type(4)))  unsigned int v4u;
typedef __attribute__((ext_vector_type(8)))  int      v8i;
typedef __attribute__((ext_vector_type(4)))  int      v4i;

#define TP 68              // f32 LDS tile pitch in floats (64 data + 4 pad dwords)
#define WP 136             // f16 W-buffer pitch in halves (128 data + 8 pad)
#define SCALE 0.125f       // d^-0.5, d=64

#if defined(__HIP_DEVICE_COMPILE__) && __has_builtin(__builtin_amdgcn_tensor_load_to_lds)
#define HAVE_TDM 1
#else
#define HAVE_TDM 0
#endif

// ---------------- WMMA helper ----------------
__device__ inline v8f wmma16(v16h a, v16h b, v8f c) {
  return __builtin_amdgcn_wmma_f32_16x16x32_f16(false, a, false, b, (short)0, c,
                                                false, false);
}

// ---------------- fragment loaders (contiguous k order, same for A and B) -----
// A (16x32 f16): lane%16 = row M; element e -> K = kb2*32 + (lane/16)*16 + e
__device__ inline v16h load_afrag_f32(const float* tile, int mbase, int kb2, int lane) {
  const int hl = lane >> 4, ln = lane & 15;
  const float* p = tile + (mbase + ln) * TP + kb2 * 32 + hl * 16;
  v16h a;
#pragma unroll
  for (int e = 0; e < 8; ++e) { a[e] = (_Float16)p[e]; a[8 + e] = (_Float16)p[8 + e]; }
  return a;
}
__device__ inline v16h load_afrag_h(const half_t* wbase, int kb2, int lane) {
  const int hl = lane >> 4, ln = lane & 15;
  const half_t* p = wbase + ln * WP + kb2 * 32 + hl * 16;
  v16h a;
#pragma unroll
  for (int e = 0; e < 8; ++e) { a[e] = p[e]; a[8 + e] = p[8 + e]; }
  return a;
}
// B (32x16 f16): lane%16 = col N; element e -> K = kb*32 + (lane/16)*16 + e
// K-tile stored row-major [n][e] -> B(k,n)=Kt[n][k] : contiguous along e
__device__ inline v16h load_bfrag_kt(const float* tile, int nbase, int kb, int lane) {
  const int hl = lane >> 4, ln = lane & 15;
  const float* p = tile + (nbase + ln) * TP + kb * 32 + hl * 16;
  v16h b;
#pragma unroll
  for (int e = 0; e < 16; ++e) b[e] = (_Float16)p[e];
  return b;
}
// V-tile stored row-major [t][e] -> B(k=t,n=e)=V[t][e] : strided along e
__device__ inline v16h load_bfrag_v(const float* tile, int nbase, int kb, int lane) {
  const int hl = lane >> 4, ln = lane & 15;
  const float* p = tile + (kb * 32 + hl * 16) * TP + nbase + ln;
  v16h b;
#pragma unroll
  for (int e = 0; e < 16; ++e) b[e] = (_Float16)p[e * TP];
  return b;
}

// ---------------- TDM: 64x64 f32 tile, global row stride 64, LDS pitch 68 ----
__device__ inline void tile_fetch(const float* gsrc, float* ldsdst, int tid) {
#if HAVE_TDM
  if (tid < 32) {   // one TDM issue from wave 0 (EXEC ignored by tensor ops)
    unsigned long long ga = (unsigned long long)gsrc;
    unsigned lo = (unsigned)(unsigned long long)ldsdst;   // LDS byte offset
    v4u g0;
    g0[0] = 1u;                                            // count = 1
    g0[1] = lo;                                            // lds_addr
    g0[2] = (unsigned)(ga & 0xFFFFFFFFull);                // global_addr lo
    g0[3] = (unsigned)((ga >> 32) & 0x1FFFFFFull) | (2u << 30); // hi | type=2
    v8i g1;
    // data_size=4B, pad_enable, pad_interval: 64 dw (code 5), pad_amount: 4 dw (code 3)
    g1[0] = (2 << 16) | (1 << 20) | (5 << 22) | (3 << 25);
    g1[1] = (int)(64u << 16);   // tensor_dim0 = 64
    g1[2] = (int)(64u << 16);   // tensor_dim1 = 64
    g1[3] = (int)(64u << 16);   // tile_dim0   = 64
    g1[4] = 64;                 // tile_dim1   = 64
    g1[5] = 64;                 // tensor_dim0_stride = 64
    g1[6] = 0;
    g1[7] = 0;
    v4i z4 = {0, 0, 0, 0};
#if __has_include(<hip/amd_detail/amd_gfx1250_TDM.h>)
    v8i z8 = {0, 0, 0, 0, 0, 0, 0, 0};
    __builtin_amdgcn_tensor_load_to_lds(g0, g1, z4, z4, z8, 0);
#else
    __builtin_amdgcn_tensor_load_to_lds(g0, g1, z4, z4, 0);
#endif
  }
#else
  for (int idx = tid; idx < 64 * 64; idx += 256) {
    const int r = idx >> 6, c = idx & 63;
    ldsdst[r * TP + c] = gsrc[r * 64 + c];
  }
#endif
}
__device__ inline void tile_wait_keep2() {   // leave the 2 newest in flight
#if HAVE_TDM
  if (threadIdx.x < 32) __builtin_amdgcn_s_wait_tensorcnt(2);
#endif
}
__device__ inline void tile_wait_all() {
#if HAVE_TDM
  if (threadIdx.x < 32) __builtin_amdgcn_s_wait_tensorcnt(0);
#endif
}

// =============== Kernel 1: Sinkhorn router -> thresholded Rm =================
__global__ __launch_bounds__(1024) void sinkhorn_r_kernel(
    const float* __restrict__ q, const float* __restrict__ k,
    const float* __restrict__ gu, float* __restrict__ Rm) {
  const int bh = blockIdx.x;               // 8
  __shared__ float qs[32 * 64];
  __shared__ float ks[32 * 64];
  __shared__ float rmat[32 * 33];
  const int tid = threadIdx.x;
  const float* qb = q + (size_t)bh * 2048 * 64;
  const float* kb = k + (size_t)bh * 2048 * 64;
  for (int idx = tid; idx < 2048; idx += 1024) {
    const int i = idx >> 6, e = idx & 63;
    const float* qp = qb + (size_t)(i * 64) * 64 + e;
    const float* kp = kb + (size_t)(i * 64) * 64 + e;
    float sq = 0.f, sk = 0.f;
    for (int r = 0; r < 64; ++r) { sq += qp[r * 64]; sk += kp[r * 64]; }
    qs[idx] = sq; ks[idx] = sk;
  }
  __syncthreads();
  const int i = tid >> 5, j = tid & 31;    // wave = row i, lane = col j
  float dot = 0.f;
  for (int e = 0; e < 64; ++e) dot += qs[i * 64 + e] * ks[j * 64 + e];
  float r = __logf(fmaxf(dot * SCALE, 0.f) + 1e-6f);
  const float u = gu[(size_t)bh * 1024 + tid];
  const float g = -__logf(-__logf(u + 1e-6f) + 1e-6f);
  r = (r + g) * (1.0f / 0.75f);
  for (int it = 0; it < 7; ++it) {
    // row logsumexp (over j = lanes of this wave)
    float m = r;
    for (int d = 16; d >= 1; d >>= 1) m = fmaxf(m, __shfl_xor(m, d, 32));
    float s = __expf(r - m);
    for (int d = 16; d >= 1; d >>= 1) s += __shfl_xor(s, d, 32);
    r -= (m + __logf(s));
    // column logsumexp (over i, via LDS)
    rmat[i * 33 + j] = r;
    __syncthreads();
    float cm = -1e30f;
    for (int ii = 0; ii < 32; ++ii) cm = fmaxf(cm, rmat[ii * 33 + j]);
    float cs = 0.f;
    for (int ii = 0; ii < 32; ++ii) cs += __expf(rmat[ii * 33 + j] - cm);
    r -= (cm + __logf(cs));
    __syncthreads();
  }
  const float R = __expf(r);
  Rm[(size_t)bh * 1024 + tid] = (R > 1e-3f) ? R : 0.f;
}

// =============== Kernel 2: gated block attention (WMMA + TDM) ================
__global__ __launch_bounds__(256) void sinkhorn_attn_kernel(
    const float* __restrict__ q, const float* __restrict__ k,
    const float* __restrict__ v, const float* __restrict__ Rm,
    float* __restrict__ out) {
  const int i   = blockIdx.x;              // bucket
  const int bh  = blockIdx.y;
  const int tid = threadIdx.x;
  const int lane = tid & 31, wave = tid >> 5;
  const int mb = wave & 3;                 // row block (16 rows)
  const int nh = wave >> 2;                // column half (32 cols)
  const int hl = lane >> 4, ln = lane & 15;

  extern __shared__ char smem_raw[];
  float*  Qt    = (float*)smem_raw;                // 64x68 f32
  float*  Vti   = Qt + 64 * TP;
  float*  Kb0   = Vti + 64 * TP;
  float*  Kb1   = Kb0 + 64 * TP;
  float*  Vb0   = Kb1 + 64 * TP;
  float*  Vb1   = Vb0 + 64 * TP;
  half_t* Wb    = (half_t*)(Vb1 + 64 * TP);        // [64 rows][136] f16
  float*  smax  = (float*)(Wb + 64 * WP);          // [64][2]
  float*  ssum  = smax + 128;                      // [64][2]
  int*    actj  = (int*)(ssum + 128);
  float*  actr  = (float*)(actj + 32);
  int*    nactp = (int*)(actr + 32);
  float* KbA[2] = {Kb0, Kb1};
  float* VbA[2] = {Vb0, Vb1};

  const size_t base = (size_t)bh * 2048 * 64;
  const float* Qg = q + base + (size_t)i * 64 * 64;
  const float* Kg = k + base;
  const float* Vg = v + base;

  if (tid == 0) {  // compact active-j list from the Sinkhorn gate
    int c = 0;
    for (int j = 0; j < 32; ++j) {
      const float rv = Rm[((size_t)bh * 32 + i) * 32 + j];
      if (rv > 0.f) { actj[c] = j; actr[c] = rv; ++c; }
    }
    *nactp = c;
  }
  tile_fetch(Qg, Qt, tid);
  tile_fetch(Vg + (size_t)i * 64 * 64, Vti, tid);
  tile_fetch(Kg + (size_t)i * 64 * 64, Kb0, tid);
  tile_wait_all();
  __syncthreads();
  const int nact = *nactp;

  // Q A-fragments stay in registers for the whole block
  const v16h Qf0 = load_afrag_f32(Qt, mb * 16, 0, lane);
  const v16h Qf1 = load_afrag_f32(Qt, mb * 16, 1, lane);

  // S_self (this wave's 16x32 slice), scaled, kept in registers
  v8f sself0, sself1;
  {
    v8f c0 = {}; v8f c1 = {};
    c0 = wmma16(Qf0, load_bfrag_kt(Kb0, (2 * nh + 0) * 16, 0, lane), c0);
    c0 = wmma16(Qf1, load_bfrag_kt(Kb0, (2 * nh + 0) * 16, 1, lane), c0);
    c1 = wmma16(Qf0, load_bfrag_kt(Kb0, (2 * nh + 1) * 16, 0, lane), c1);
    c1 = wmma16(Qf1, load_bfrag_kt(Kb0, (2 * nh + 1) * 16, 1, lane), c1);
    sself0 = c0 * SCALE; sself1 = c1 * SCALE;
  }
  __syncthreads();  // Kb0 free for streaming

  if (nact > 0) {
    tile_fetch(Kg + (size_t)actj[0] * 64 * 64, Kb0, tid);
    tile_fetch(Vg + (size_t)actj[0] * 64 * 64, Vb0, tid);
  }

  v8f oacc0 = {}; v8f oacc1 = {};
  for (int a = 0; a < nact; ++a) {
    const int cur = a & 1;
    if (a + 1 < nact) {  // double-buffered TDM prefetch of next active tiles
      tile_fetch(Kg + (size_t)actj[a + 1] * 64 * 64, KbA[cur ^ 1], tid);
      tile_fetch(Vg + (size_t)actj[a + 1] * 64 * 64, VbA[cur ^ 1], tid);
      tile_wait_keep2();
    } else {
      tile_wait_all();
    }
    __syncthreads();
    const float rm = actr[a];
    const float* Kc = KbA[cur];
    const float* Vc = VbA[cur];

    // S_cross = Qi * Kj^T (this wave's 16x32 slice)
    v8f sc0, sc1;
    {
      v8f c0 = {}; v8f c1 = {};
      c0 = wmma16(Qf0, load_bfrag_kt(Kc, (2 * nh + 0) * 16, 0, lane), c0);
      c0 = wmma16(Qf1, load_bfrag_kt(Kc, (2 * nh + 0) * 16, 1, lane), c0);
      c1 = wmma16(Qf0, load_bfrag_kt(Kc, (2 * nh + 1) * 16, 0, lane), c1);
      c1 = wmma16(Qf1, load_bfrag_kt(Kc, (2 * nh + 1) * 16, 1, lane), c1);
      sc0 = c0 * SCALE; sc1 = c1 * SCALE;
    }

    // row softmax over 128 = [self(64) | cross(64)], rows split across 2 waves
    float pm[8];
#pragma unroll
    for (int r = 0; r < 8; ++r) {
      float m = fmaxf(fmaxf(sself0[r], sself1[r]), fmaxf(sc0[r], sc1[r]));
      m = fmaxf(m, __shfl_xor(m, 8, 32));
      m = fmaxf(m, __shfl_xor(m, 4, 32));
      m = fmaxf(m, __shfl_xor(m, 2, 32));
      m = fmaxf(m, __shfl_xor(m, 1, 32));
      pm[r] = m;
    }
    if (ln == 0) {
#pragma unroll
      for (int r = 0; r < 8; ++r) smax[(mb * 16 + hl * 8 + r) * 2 + nh] = pm[r];
    }
    __syncthreads();
    float ex0[8], ex1[8], ex2[8], ex3[8], inv[8];
#pragma unroll
    for (int r = 0; r < 8; ++r) {
      const int row = mb * 16 + hl * 8 + r;
      const float rmax = fmaxf(smax[row * 2 + 0], smax[row * 2 + 1]);
      ex0[r] = __expf(sself0[r] - rmax);
      ex1[r] = __expf(sself1[r] - rmax);
      ex2[r] = __expf(sc0[r] - rmax);
      ex3[r] = __expf(sc1[r] - rmax);
      float s = ex0[r] + ex1[r] + ex2[r] + ex3[r];
      s += __shfl_xor(s, 8, 32);
      s += __shfl_xor(s, 4, 32);
      s += __shfl_xor(s, 2, 32);
      s += __shfl_xor(s, 1, 32);
      inv[r] = s;                               // partial (this wave's 64 cols)
    }
    if (ln == 0) {
#pragma unroll
      for (int r = 0; r < 8; ++r) ssum[(mb * 16 + hl * 8 + r) * 2 + nh] = inv[r];
    }
    __syncthreads();
#pragma unroll
    for (int r = 0; r < 8; ++r) {
      const int row = mb * 16 + hl * 8 + r;
      inv[r] = rm / (ssum[row * 2 + 0] + ssum[row * 2 + 1]);  // fold Rm gate in
    }
    // C-layout -> f16 A-layout staging of W = softmax * Rm
    {
      const int c0 = 2 * nh * 16 + ln;
#pragma unroll
      for (int r = 0; r < 8; ++r) {
        half_t* wr = Wb + (mb * 16 + hl * 8 + r) * WP;
        wr[c0]           = (half_t)(ex0[r] * inv[r]);   // self cols
        wr[c0 + 16]      = (half_t)(ex1[r] * inv[r]);
        wr[64 + c0]      = (half_t)(ex2[r] * inv[r]);   // cross cols
        wr[64 + c0 + 16] = (half_t)(ex3[r] * inv[r]);
      }
    }
    __syncthreads();

    // out += W_self * Vi + W_cross * Vj  (K-dim = 128)
    {
      const half_t* wrow = Wb + mb * 16 * WP;
      const v16h wa0 = load_afrag_h(wrow, 0, lane);
      const v16h wa1 = load_afrag_h(wrow, 1, lane);
      const v16h wa2 = load_afrag_h(wrow, 2, lane);
      const v16h wa3 = load_afrag_h(wrow, 3, lane);
      v8f c0 = oacc0, c1 = oacc1;
      c0 = wmma16(wa0, load_bfrag_v(Vti, (2 * nh + 0) * 16, 0, lane), c0);
      c0 = wmma16(wa1, load_bfrag_v(Vti, (2 * nh + 0) * 16, 1, lane), c0);
      c0 = wmma16(wa2, load_bfrag_v(Vc,  (2 * nh + 0) * 16, 0, lane), c0);
      c0 = wmma16(wa3, load_bfrag_v(Vc,  (2 * nh + 0) * 16, 1, lane), c0);
      c1 = wmma16(wa0, load_bfrag_v(Vti, (2 * nh + 1) * 16, 0, lane), c1);
      c1 = wmma16(wa1, load_bfrag_v(Vti, (2 * nh + 1) * 16, 1, lane), c1);
      c1 = wmma16(wa2, load_bfrag_v(Vc,  (2 * nh + 1) * 16, 0, lane), c1);
      c1 = wmma16(wa3, load_bfrag_v(Vc,  (2 * nh + 1) * 16, 1, lane), c1);
      oacc0 = c0; oacc1 = c1;
    }
    __syncthreads();  // protect Wb / K,V buffers before next iteration
  }

  // epilogue: C-fragment -> global (f32), lanes 0-15 write consecutive cols
  const size_t obase = base + (size_t)i * 64 * 64;
#pragma unroll
  for (int r = 0; r < 8; ++r) {
    const int row = mb * 16 + hl * 8 + r;
    out[obase + (size_t)row * 64 + (2 * nh + 0) * 16 + ln] = oacc0[r];
    out[obase + (size_t)row * 64 + (2 * nh + 1) * 16 + ln] = oacc1[r];
  }
}

// ============================= host launcher =================================
extern "C" void kernel_launch(void* const* d_in, const int* in_sizes, int n_in,
                              void* d_out, int out_size, void* d_ws, size_t ws_size,
                              hipStream_t stream) {
  const float* q  = (const float*)d_in[0];
  const float* k  = (const float*)d_in[1];
  const float* v  = (const float*)d_in[2];
  const float* gu = (const float*)d_in[3];
  float* Rm  = (float*)d_ws;               // 8*32*32 f32 gates
  float* out = (float*)d_out;

  sinkhorn_r_kernel<<<8, 1024, 0, stream>>>(q, k, gu, Rm);

  const size_t smem = (size_t)(64 * TP) * 6 * sizeof(float)   // Q,Vi,K0,K1,V0,V1
                    + (size_t)64 * WP * sizeof(half_t)        // W buffer
                    + 256 * sizeof(float)                     // smax+ssum
                    + 32 * sizeof(int) + 32 * sizeof(float) + 16;
  (void)hipFuncSetAttribute((const void*)sinkhorn_attn_kernel,
                            hipFuncAttributeMaxDynamicSharedMemorySize, (int)smem);
  dim3 grid(32, 8);
  sinkhorn_attn_kernel<<<grid, 256, smem, stream>>>(q, k, v, Rm, out);
}
